// DistributedDotGAT_19542101196806
// MI455X (gfx1250) — matrix operand order repaired and checked by reading, not compiled, not run on hardware
//
#include <hip/hip_runtime.h>
#include <hip/hip_bf16.h>
#include <math.h>

typedef __attribute__((ext_vector_type(16))) _Float16 v16h;
typedef __attribute__((ext_vector_type(8)))  float    v8f;

#define B_   32
#define A_   100
#define D_   10000
#define HID  256
#define NH   4
#define OUT_ 10000
#define NFREQ 16
#define E_   100
#define NSQ  100        // sqrt(D)
#define KF   64         // feature dim 1+2*16=33 padded to 64
#define AP   112        // agents padded to 7*16
#define APK  128        // K-pad for alpha@V
#define TWO_PI 6.28318530717958647692f

__device__ __forceinline__ int laneid() { return threadIdx.x & 31; }
__device__ __forceinline__ int waveid() { return threadIdx.x >> 5; }
__device__ __forceinline__ float swishf(float v) { return v / (1.0f + expf(-v)); }

// ---- WMMA fragment helpers (CDNA5 16x16x32 f16, wave32; contiguous k = 16*(lane/16)+s in both A and B) ----

// A fragment: 16x32 f16 tile, row-major source, leading dim ld.
// lanes 0-15: M=lane, K 0..15; lanes 16-31: M=lane-16, K 16..31
__device__ __forceinline__ v16h load_a_frag(const _Float16* base, int ld) {
    int lane = laneid();
    int m = lane & 15;
    int koff = (lane >> 4) << 4;
    v16h a;
#pragma unroll
    for (int h = 0; h < 16; ++h) {
        int k = h + koff;
        a[h] = base[m * ld + k];
    }
    return a;
}

// B fragment from [N][K] row-major (native weight layout W[out][in]):
// lanes 0-15: N=lane, K 0..15; lanes 16-31: N=lane-16, K 16..31  (contiguous 32B/lane)
__device__ __forceinline__ v16h load_b_frag_nk(const _Float16* base, int ld) {
    int lane = laneid();
    int n = lane & 15;
    int koff = (lane >> 4) << 4;
    v16h b;
#pragma unroll
    for (int h = 0; h < 16; ++h) b[h] = base[n * ld + koff + h];
    return b;
}

// Prefetch the cacheline this lane will need for a future B fragment.
// Locality 3 -> WGP-scope prefetch: pulls into all cache levels (ISA 10.5),
// hiding latency for the subsequent global_load_b128; does not touch LOADcnt.
__device__ __forceinline__ void prefetch_b_nk(const _Float16* base, int ld) {
    int lane = laneid();
    __builtin_prefetch((const void*)(base + (size_t)(lane & 15) * ld + ((lane >> 4) << 4)), 0, 3);
}

// B fragment from [K][N] row-major (e.g. V matrix [agent][hid])
__device__ __forceinline__ v16h load_b_frag_kn(const _Float16* base, int ld) {
    int lane = laneid();
    int n = lane & 15;
    int koff = (lane >> 4) << 4;
    v16h b;
#pragma unroll
    for (int h = 0; h < 16; ++h) b[h] = base[(koff + h) * ld + n];
    return b;
}

#define WMMA_F16(a, b, c) \
    __builtin_amdgcn_wmma_f32_16x16x32_f16(false, (a), false, (b), (short)0, (c), false, false)

// ---------------- weight conversion f32 -> f16 with optional K padding ----------------
__global__ void cvt_pad_kernel(const float* __restrict__ src, _Float16* __restrict__ dst,
                               int rows, int sk, int dk) {
    size_t n = (size_t)rows * dk;
    for (size_t i = blockIdx.x * (size_t)blockDim.x + threadIdx.x; i < n;
         i += (size_t)gridDim.x * blockDim.x) {
        int r = (int)(i / dk), k = (int)(i % dk);
        dst[i] = (k < sk) ? (_Float16)src[(size_t)r * sk + k] : (_Float16)0.0f;
    }
}

// ---------------- gather nonzeros (ascending flat index) + Fourier features ----------------
__global__ void __launch_bounds__(256) gather_feat_kernel(
    const float* __restrict__ x, const float* __restrict__ fB,
    _Float16* __restrict__ feat, unsigned char* __restrict__ valid) {
    __shared__ int s_cnt[256];
    __shared__ int s_idx[E_];
    __shared__ int s_total;
    int ba = blockIdx.x;
    const float* xr = x + (size_t)ba * D_;
    int t = threadIdx.x;
    const int CH = (D_ + 255) / 256;          // 40
    int st = t * CH;
    int en = st + CH < D_ ? st + CH : D_;
    int c = 0;
    for (int d = st; d < en; ++d) c += (xr[d] != 0.0f) ? 1 : 0;
    s_cnt[t] = c;
    __syncthreads();
    if (t == 0) {                              // exclusive prefix sum (chunk order == index order)
        int run = 0;
        for (int i = 0; i < 256; ++i) { int v = s_cnt[i]; s_cnt[i] = run; run += v; }
        s_total = run;
    }
    __syncthreads();
    int off = s_cnt[t];
    for (int d = st; d < en; ++d) {
        if (xr[d] != 0.0f) { if (off < E_) s_idx[off] = d; ++off; }
    }
    __syncthreads();
    int nv = s_total < E_ ? s_total : E_;
    for (int j = t; j < E_; j += 256) {
        _Float16* fr = feat + ((size_t)ba * E_ + j) * KF;
        bool ok = j < nv;
        valid[(size_t)ba * E_ + j] = ok ? 1 : 0;
        float val = 0.f, row = 0.f, col = 0.f;
        if (ok) {
            int d = s_idx[j];
            val = xr[d];
            row = (float)(d / NSQ);
            col = (float)(d % NSQ);
        }
        fr[0] = (_Float16)val;
#pragma unroll
        for (int f = 0; f < NFREQ; ++f) {
            float p = TWO_PI * (row * fB[2 * f] + col * fB[2 * f + 1]);
            fr[1 + f]         = (_Float16)(ok ? sinf(p) : 0.f);
            fr[1 + NFREQ + f] = (_Float16)(ok ? cosf(p) : 0.f);
        }
        for (int k = 1 + 2 * NFREQ; k < KF; ++k) fr[k] = (_Float16)0.f;
    }
}

// ------- fused entry-encoder MLP + compression GEMM ([3200 x 25600] @ [25600 x 512]) -------
// Block = 16 (b,a) rows, 8 waves. e never hits HBM; compression acc lives in VGPRs.
__global__ void __launch_bounds__(256) enc_comp_kernel(
    const _Float16* __restrict__ feat, const unsigned char* __restrict__ valid,
    const _Float16* __restrict__ W1t, const float* __restrict__ b1,
    const _Float16* __restrict__ W2t, const float* __restrict__ b2,
    const float* __restrict__ pad_tok,
    const _Float16* __restrict__ cW1t, const float* __restrict__ cb1,
    const _Float16* __restrict__ cW2t, const float* __restrict__ cb2,
    float* __restrict__ h32, _Float16* __restrict__ h16) {
    __shared__ _Float16 sA[16 * KF];
    __shared__ _Float16 sH1[16 * HID];
    __shared__ _Float16 sE[16 * HID];
    __shared__ _Float16 sHf[16 * 512];
    int row0 = blockIdx.x * 16;
    int t = threadIdx.x, w = waveid(), lane = laneid();
    int mloc = (lane >> 4) << 3, nloc = lane & 15;
    v8f vzero = {};
    v8f acc1[4];                               // persistent 16x512 compression accumulators
#pragma unroll
    for (int i = 0; i < 4; ++i) acc1[i] = vzero;

    for (int j = 0; j < E_; ++j) {
        for (int i = t; i < 16 * KF; i += 256) {
            int m = i / KF, k = i % KF;
            sA[i] = feat[((size_t)(row0 + m) * E_ + j) * KF + k];
            // pull next entry's feature tile toward the caches early
            if (k == 0)
                __builtin_prefetch(
                    (const void*)&feat[((size_t)(row0 + m) * E_ + j + 1) * KF], 0, 3);
        }
        __syncthreads();
        // H1 = swish(feat @ enc_W1^T + b1) : K=64, N=256
#pragma unroll
        for (int ii = 0; ii < 2; ++ii) {
            int nt = w * 2 + ii;
            v8f acc = vzero;
#pragma unroll
            for (int ks = 0; ks < KF / 32; ++ks) {
                v16h a  = load_a_frag(sA + ks * 32, KF);
                v16h bf = load_b_frag_nk(W1t + (size_t)(nt * 16) * KF + ks * 32, KF);
                acc = WMMA_F16(a, bf, acc);
            }
#pragma unroll
            for (int r = 0; r < 8; ++r) {
                int m = mloc + r, n = nt * 16 + nloc;
                sH1[m * HID + n] = (_Float16)swishf(acc[r] + b1[n]);
            }
        }
        __syncthreads();
        // E = H1 @ enc_W2^T + b2 ; invalid slots -> pad_token
#pragma unroll
        for (int ii = 0; ii < 2; ++ii) {
            int nt = w * 2 + ii;
            v8f acc = vzero;
#pragma unroll
            for (int ks = 0; ks < HID / 32; ++ks) {
                v16h a  = load_a_frag(sH1 + ks * 32, HID);
                v16h bf = load_b_frag_nk(W2t + (size_t)(nt * 16) * HID + ks * 32, HID);
                acc = WMMA_F16(a, bf, acc);
            }
#pragma unroll
            for (int r = 0; r < 8; ++r) {
                int m = mloc + r, n = nt * 16 + nloc;
                float v = acc[r] + b2[n];
                bool ok = valid[(size_t)(row0 + m) * E_ + j] != 0;
                sE[m * HID + n] = (_Float16)(ok ? v : pad_tok[n]);
            }
        }
        __syncthreads();
        // compression partial: acc1 += E_j @ comp_W1^T[:, j*256 : j*256+256]
#pragma unroll
        for (int ii = 0; ii < 4; ++ii) {
            int nt = w * 4 + ii;
#pragma unroll
            for (int ks = 0; ks < HID / 32; ++ks) {
                const _Float16* bbase =
                    cW1t + (size_t)(nt * 16) * (E_ * HID) + j * HID + ks * 32;
                v16h a  = load_a_frag(sE + ks * 32, HID);
                v16h bf = load_b_frag_nk(bbase, E_ * HID);
                // prefetch next j's slice of the same weight rows (+HID columns)
                prefetch_b_nk(bbase + HID, E_ * HID);
                acc1[ii] = WMMA_F16(a, bf, acc1[ii]);
            }
        }
        __syncthreads();
    }
    // swish(acc1 + comp_b1) -> [16][512]
#pragma unroll
    for (int ii = 0; ii < 4; ++ii) {
        int nt = w * 4 + ii;
#pragma unroll
        for (int r = 0; r < 8; ++r) {
            int m = mloc + r, n = nt * 16 + nloc;
            sHf[m * 512 + n] = (_Float16)swishf(acc1[ii][r] + cb1[n]);
        }
    }
    __syncthreads();
    // h = sHf @ comp_W2^T + comp_b2 : K=512, N=256
#pragma unroll
    for (int ii = 0; ii < 2; ++ii) {
        int nt = w * 2 + ii;
        v8f acc = vzero;
#pragma unroll
        for (int ks = 0; ks < 512 / 32; ++ks) {
            v16h a  = load_a_frag(sHf + ks * 32, 512);
            v16h bf = load_b_frag_nk(cW2t + (size_t)(nt * 16) * 512 + ks * 32, 512);
            acc = WMMA_F16(a, bf, acc);
        }
#pragma unroll
        for (int r = 0; r < 8; ++r) {
            int m = mloc + r, n = nt * 16 + nloc;
            float v = acc[r] + cb2[n];
            h32[(size_t)(row0 + m) * HID + n] = v;
            h16[(size_t)(row0 + m) * HID + n] = (_Float16)v;
        }
    }
}

// ---------------- one GAT step, one (batch, head) per block; all tiles in LDS ----------------
__global__ void __launch_bounds__(256) gat_head_kernel(
    const _Float16* __restrict__ h16, const float* __restrict__ conn,
    const _Float16* __restrict__ qWt, const _Float16* __restrict__ kWt,
    const _Float16* __restrict__ vWt,
    const _Float16* __restrict__ fW1t, const float* __restrict__ fb1,
    const _Float16* __restrict__ fW2t, const float* __restrict__ fb2,
    const float* __restrict__ lng, const float* __restrict__ lnb,
    float* __restrict__ oph, float inv_scale) {
    extern __shared__ char smem[];
    _Float16* sH  = (_Float16*)smem;               // AP x HID   (later: swish(Hm))
    _Float16* sQ  = sH + AP * HID;                 // AP x HID   (later: swish(t1))
    _Float16* sK  = sQ + AP * HID;                 // AP x HID   (later: t2)
    _Float16* sV  = sK + AP * HID;                 // APK x HID  (rows 112..127 zero)
    float*    sS  = (float*)(sV + (size_t)APK * HID);   // AP x AP scores
    _Float16* sAl = (_Float16*)(sS + (size_t)AP * AP);  // AP x APK alpha
    int b  = blockIdx.x >> 2;
    int hd = blockIdx.x & 3;
    int t = threadIdx.x, w = waveid(), lane = laneid();
    int mloc = (lane >> 4) << 3, nloc = lane & 15;
    v8f vzero = {};

    for (int i = t; i < AP * HID; i += 256) {
        int m = i / HID, k = i % HID;
        sH[i] = (m < A_) ? h16[((size_t)b * A_ + m) * HID + k] : (_Float16)0.f;
    }
    for (int i = t; i < (APK - AP) * HID; i += 256) sV[AP * HID + i] = (_Float16)0.f;
    __syncthreads();

    // Q (pre-scaled by 1/sqrt(HID)), K, V : 3 * 7 * 16 tiles
    const _Float16* Wq = qWt + (size_t)hd * HID * HID;
    const _Float16* Wk = kWt + (size_t)hd * HID * HID;
    const _Float16* Wv = vWt + (size_t)hd * HID * HID;
    for (int job = w; job < 3 * 7 * 16; job += 8) {
        int which = job / 112, tile = job % 112;
        int mt = tile / 16, nt = tile % 16;
        const _Float16* W = (which == 0) ? Wq : (which == 1 ? Wk : Wv);
        _Float16* dst     = (which == 0) ? sQ : (which == 1 ? sK : sV);
        float scl         = (which == 0) ? inv_scale : 1.0f;
        v8f acc = vzero;
#pragma unroll
        for (int ks = 0; ks < 8; ++ks) {
            v16h a  = load_a_frag(sH + mt * 16 * HID + ks * 32, HID);
            v16h bf = load_b_frag_nk(W + (size_t)(nt * 16) * HID + ks * 32, HID);
            acc = WMMA_F16(a, bf, acc);
        }
#pragma unroll
        for (int r = 0; r < 8; ++r)
            dst[(mt * 16 + mloc + r) * HID + nt * 16 + nloc] = (_Float16)(acc[r] * scl);
    }
    __syncthreads();

    // scores = Q @ K^T + conn  (K stored [agent][hid] == [N][K] for the B reader)
    for (int tile = w; tile < 49; tile += 8) {
        int mt = tile / 7, nt = tile % 7;
        v8f acc = vzero;
#pragma unroll
        for (int ks = 0; ks < 8; ++ks) {
            v16h a  = load_a_frag(sQ + mt * 16 * HID + ks * 32, HID);
            v16h bf = load_b_frag_nk(sK + nt * 16 * HID + ks * 32, HID);
            acc = WMMA_F16(a, bf, acc);
        }
#pragma unroll
        for (int r = 0; r < 8; ++r) {
            int ar = mt * 16 + mloc + r, cc = nt * 16 + nloc;
            float s = (cc >= A_) ? -1e30f
                                 : acc[r] + ((ar < A_) ? conn[ar * A_ + cc] : 0.f);
            sS[ar * AP + cc] = s;
        }
    }
    __syncthreads();

    // row softmax -> alpha (f16, K padded to 128)
    for (int r = t; r < AP; r += 256) {
        float mx = -1e30f;
        for (int c2 = 0; c2 < AP; ++c2) mx = fmaxf(mx, sS[r * AP + c2]);
        float sum = 0.f;
        for (int c2 = 0; c2 < AP; ++c2) sum += expf(sS[r * AP + c2] - mx);
        float inv = 1.0f / sum;
        for (int c2 = 0; c2 < APK; ++c2)
            sAl[r * APK + c2] =
                (_Float16)((c2 < AP) ? expf(sS[r * AP + c2] - mx) * inv : 0.f);
    }
    __syncthreads();

    // Hm = alpha @ V ; store swish(Hm) into sH
    for (int tile = w; tile < 112; tile += 8) {
        int mt = tile / 16, nt = tile % 16;
        v8f acc = vzero;
#pragma unroll
        for (int ks = 0; ks < APK / 32; ++ks) {
            v16h a  = load_a_frag(sAl + mt * 16 * APK + ks * 32, APK);
            v16h bf = load_b_frag_kn(sV + (size_t)(ks * 32) * HID + nt * 16, HID);
            acc = WMMA_F16(a, bf, acc);
        }
#pragma unroll
        for (int r = 0; r < 8; ++r)
            sH[(mt * 16 + mloc + r) * HID + nt * 16 + nloc] = (_Float16)swishf(acc[r]);
    }
    __syncthreads();

    // t1 = swish( swish(Hm) @ fp_W1^T + b1 ) -> sQ
    const _Float16* W1 = fW1t + (size_t)hd * HID * HID;
    const float* bb1 = fb1 + hd * HID;
    for (int tile = w; tile < 112; tile += 8) {
        int mt = tile / 16, nt = tile % 16;
        v8f acc = vzero;
#pragma unroll
        for (int ks = 0; ks < 8; ++ks) {
            v16h a  = load_a_frag(sH + mt * 16 * HID + ks * 32, HID);
            v16h bf = load_b_frag_nk(W1 + (size_t)(nt * 16) * HID + ks * 32, HID);
            acc = WMMA_F16(a, bf, acc);
        }
#pragma unroll
        for (int r = 0; r < 8; ++r) {
            int n = nt * 16 + nloc;
            sQ[(mt * 16 + mloc + r) * HID + n] = (_Float16)swishf(acc[r] + bb1[n]);
        }
    }
    __syncthreads();

    // t2 = t1 @ fp_W2^T + b2 -> sK
    const _Float16* W2 = fW2t + (size_t)hd * HID * HID;
    const float* bb2 = fb2 + hd * HID;
    for (int tile = w; tile < 112; tile += 8) {
        int mt = tile / 16, nt = tile % 16;
        v8f acc = vzero;
#pragma unroll
        for (int ks = 0; ks < 8; ++ks) {
            v16h a  = load_a_frag(sQ + mt * 16 * HID + ks * 32, HID);
            v16h bf = load_b_frag_nk(W2 + (size_t)(nt * 16) * HID + ks * 32, HID);
            acc = WMMA_F16(a, bf, acc);
        }
#pragma unroll
        for (int r = 0; r < 8; ++r) {
            int n = nt * 16 + nloc;
            sK[(mt * 16 + mloc + r) * HID + n] = (_Float16)(acc[r] + bb2[n]);
        }
    }
    __syncthreads();

    // layernorm + write per-head output
    const float* g  = lng + hd * HID;
    const float* be = lnb + hd * HID;
    for (int r = t; r < A_; r += 256) {
        float mean = 0.f;
        for (int c2 = 0; c2 < HID; ++c2) mean += (float)sK[r * HID + c2];
        mean *= (1.0f / HID);
        float var = 0.f;
        for (int c2 = 0; c2 < HID; ++c2) {
            float d = (float)sK[r * HID + c2] - mean;
            var += d * d;
        }
        var *= (1.0f / HID);
        float rstd = rsqrtf(var + 1e-5f);
        float* orow = oph + ((size_t)hd * B_ * A_ + (size_t)b * A_ + r) * HID;
        for (int c2 = 0; c2 < HID; ++c2)
            orow[c2] = ((float)sK[r * HID + c2] - mean) * rstd * g[c2] + be[c2];
    }
}

// ---------------- mean over heads -> next h (f32 + f16) ----------------
__global__ void head_mean_kernel(const float* __restrict__ oph, float* __restrict__ h32,
                                 _Float16* __restrict__ h16, size_t n) {
    size_t S = n;   // stride between heads
    for (size_t i = blockIdx.x * (size_t)blockDim.x + threadIdx.x; i < n;
         i += (size_t)gridDim.x * blockDim.x) {
        float v = (oph[i] + oph[i + S] + oph[i + 2 * S] + oph[i + 3 * S]) * 0.25f;
        h32[i] = v;
        h16[i] = (_Float16)v;
    }
}

// ---------------- output projection [3200 x 256] @ [256 x 10000] ----------------
__global__ void __launch_bounds__(256) out_proj_kernel(
    const _Float16* __restrict__ h16, const _Float16* __restrict__ oWt,
    const float* __restrict__ ob, float* __restrict__ out) {
    __shared__ _Float16 sA[16 * HID];
    int row0 = blockIdx.x * 16;
    int nt0  = blockIdx.y * 16;
    int t = threadIdx.x, w = waveid(), lane = laneid();
    int mloc = (lane >> 4) << 3, nloc = lane & 15;
    for (int i = t; i < 16 * HID; i += 256) sA[i] = h16[(size_t)row0 * HID + i];
    __syncthreads();
    v8f vzero = {};
    const int NTILES = OUT_ / 16;   // 625
    for (int i = w; i < 16; i += 8) {
        int nt = nt0 + i;
        if (nt >= NTILES) continue;
        v8f acc = vzero;
#pragma unroll
        for (int ks = 0; ks < 8; ++ks) {
            v16h a  = load_a_frag(sA + ks * 32, HID);
            v16h bf = load_b_frag_nk(oWt + (size_t)(nt * 16) * HID + ks * 32, HID);
            acc = WMMA_F16(a, bf, acc);
        }
#pragma unroll
        for (int r = 0; r < 8; ++r) {
            int m = row0 + mloc + r, n = nt * 16 + nloc;
            out[(size_t)m * OUT_ + n] = acc[r] + ob[n];
        }
    }
}

extern "C" void kernel_launch(void* const* d_in, const int* in_sizes, int n_in,
                              void* d_out, int out_size, void* d_ws, size_t ws_size,
                              hipStream_t stream) {
    (void)in_sizes; (void)n_in; (void)out_size; (void)ws_size;
    const float* x      = (const float*)d_in[0];
    const float* fB     = (const float*)d_in[1];
    const float* encW1  = (const float*)d_in[2];
    const float* encb1  = (const float*)d_in[3];
    const float* encW2  = (const float*)d_in[4];
    const float* encb2  = (const float*)d_in[5];
    const float* padtok = (const float*)d_in[6];
    const float* compW1 = (const float*)d_in[7];
    const float* compb1 = (const float*)d_in[8];
    const float* compW2 = (const float*)d_in[9];
    const float* compb2 = (const float*)d_in[10];
    const float* conn   = (const float*)d_in[11];
    const float* qW     = (const float*)d_in[12];
    const float* kW     = (const float*)d_in[13];
    const float* vW     = (const float*)d_in[14];
    const float* fpW1   = (const float*)d_in[15];
    const float* fpb1   = (const float*)d_in[16];
    const float* fpW2   = (const float*)d_in[17];
    const float* fpb2   = (const float*)d_in[18];
    const float* lng    = (const float*)d_in[19];
    const float* lnb    = (const float*)d_in[20];
    const float* outW   = (const float*)d_in[21];
    const float* outb   = (const float*)d_in[22];
    float* out = (float*)d_out;

    const int ROWS = B_ * A_;   // 3200
    char* wsp = (char*)d_ws;
    auto alloc = [&](size_t bytes) -> char* {
        char* p = wsp;
        wsp += (bytes + 255) & ~(size_t)255;
        return p;
    };
    _Float16* feat    = (_Float16*)alloc((size_t)ROWS * E_ * KF * 2);
    unsigned char* vld= (unsigned char*)alloc((size_t)ROWS * E_);
    _Float16* encW1t  = (_Float16*)alloc((size_t)HID * KF * 2);
    _Float16* encW2t  = (_Float16*)alloc((size_t)HID * HID * 2);
    _Float16* compW1t = (_Float16*)alloc((size_t)512 * E_ * HID * 2);
    _Float16* compW2t = (_Float16*)alloc((size_t)HID * 512 * 2);
    _Float16* qWt     = (_Float16*)alloc((size_t)NH * HID * HID * 2);
    _Float16* kWt     = (_Float16*)alloc((size_t)NH * HID * HID * 2);
    _Float16* vWt     = (_Float16*)alloc((size_t)NH * HID * HID * 2);
    _Float16* fpW1t   = (_Float16*)alloc((size_t)NH * HID * HID * 2);
    _Float16* fpW2t   = (_Float16*)alloc((size_t)NH * HID * HID * 2);
    _Float16* outWt   = (_Float16*)alloc((size_t)OUT_ * HID * 2);
    float*    h32     = (float*)alloc((size_t)ROWS * HID * 4);
    _Float16* h16     = (_Float16*)alloc((size_t)ROWS * HID * 2);
    float*    oph     = (float*)alloc((size_t)NH * ROWS * HID * 4);

    dim3 blk(256);
    // one-time per-call weight conversions (all land in the 192MB L2)
    cvt_pad_kernel<<<64,   blk, 0, stream>>>(encW1,  encW1t,  HID, 33,  KF);
    cvt_pad_kernel<<<128,  blk, 0, stream>>>(encW2,  encW2t,  HID, HID, HID);
    cvt_pad_kernel<<<4096, blk, 0, stream>>>(compW1, compW1t, 512, E_ * HID, E_ * HID);
    cvt_pad_kernel<<<256,  blk, 0, stream>>>(compW2, compW2t, HID, 512, 512);
    cvt_pad_kernel<<<512,  blk, 0, stream>>>(qW,   qWt,   NH * HID, HID, HID);
    cvt_pad_kernel<<<512,  blk, 0, stream>>>(kW,   kWt,   NH * HID, HID, HID);
    cvt_pad_kernel<<<512,  blk, 0, stream>>>(vW,   vWt,   NH * HID, HID, HID);
    cvt_pad_kernel<<<512,  blk, 0, stream>>>(fpW1, fpW1t, NH * HID, HID, HID);
    cvt_pad_kernel<<<512,  blk, 0, stream>>>(fpW2, fpW2t, NH * HID, HID, HID);
    cvt_pad_kernel<<<2048, blk, 0, stream>>>(outW, outWt, OUT_, HID, HID);

    gather_feat_kernel<<<ROWS, blk, 0, stream>>>(x, fB, feat, vld);

    enc_comp_kernel<<<ROWS / 16, blk, 0, stream>>>(feat, vld, encW1t, encb1, encW2t, encb2,
                                                   padtok, compW1t, compb1, compW2t, compb2,
                                                   h32, h16);

    size_t gat_smem = (size_t)(AP * HID * 2) * 3 + (size_t)APK * HID * 2 +
                      (size_t)AP * AP * 4 + (size_t)AP * APK * 2;   // ~316 KB (CDNA5 LDS)
    hipFuncSetAttribute((const void*)gat_head_kernel,
                        hipFuncAttributeMaxDynamicSharedMemorySize, (int)gat_smem);
    float inv_scale = 1.0f / sqrtf((float)HID);
    for (int s = 0; s < 3; ++s) {
        gat_head_kernel<<<B_ * NH, blk, gat_smem, stream>>>(
            h16, conn, qWt, kWt, vWt, fpW1t, fpb1, fpW2t, fpb2, lng, lnb, oph, inv_scale);
        head_mean_kernel<<<1024, blk, 0, stream>>>(oph, h32, h16, (size_t)ROWS * HID);
    }

    out_proj_kernel<<<dim3(ROWS / 16, (OUT_ / 16 + 15) / 16), blk, 0, stream>>>(
        h16, outWt, outb, out);
}
